// LocalGCNEncoder_88768384073941
// MI455X (gfx1250) — matrix operand order repaired and checked by reading, not compiled, not run on hardware
//
#include <hip/hip_runtime.h>

typedef __attribute__((ext_vector_type(16))) _Float16 v16h;
typedef __attribute__((ext_vector_type(8)))  _Float16 h8;
typedef __attribute__((ext_vector_type(8)))  float    v8f;

#define DEVINL __device__ __forceinline__

static constexpr int NN = 50000;   // nodes
static constexpr int NE = 800000;  // edges
static constexpr int NG = 64;      // graphs

DEVINL h8 ld8(const _Float16* p) { return *(const h8*)p; }

DEVINL v16h mk16(h8 a, h8 b) {
  return __builtin_shufflevector(a, b, 0,1,2,3,4,5,6,7,8,9,10,11,12,13,14,15);
}

DEVINL v8f vzero() { v8f z = {0.f,0.f,0.f,0.f,0.f,0.f,0.f,0.f}; return z; }

// monotonic float <-> uint encoding so atomicMax(uint) == float max
DEVINL unsigned encf(float f) {
  unsigned s = __float_as_uint(f);
  return s ^ (unsigned)(((int)s >> 31) | 0x80000000u);
}
DEVINL float decf(unsigned u) {
  unsigned m = (unsigned)(((int)(~u) >> 31) | 0x80000000u);
  return __uint_as_float(u ^ m);
}

// ---------------- generic WMMA GEMM: C[M,ldC] = A[M,K](f16) @ Bt[N,K](f16)^T -------------
// one wave -> 16 rows x 128 cols; grid.y selects which 128-col slab.
template<int K>
__global__ void gemm_f16(const _Float16* __restrict__ A, const _Float16* __restrict__ Bt,
                         float* __restrict__ C, int rowTiles, int ldC) {
  int wave = blockIdx.x * (blockDim.x >> 5) + (threadIdx.x >> 5);
  if (wave >= rowTiles) return;                 // wave-uniform guard (EXEC all-ones inside)
  int lane = threadIdx.x & 31;
  int arow = lane & 15;                         // A: row in tile / B: col in tile
  int akb  = (lane < 16) ? 0 : 8;               // A-operand K sub-chunk base
  int bkb  = (lane < 16) ? 0 : 8;               // B-operand K sub-chunk base
  int ntOff = blockIdx.y * 8;

  const _Float16* ap = A + (size_t)(wave * 16 + arow) * K;

  v8f acc[8];
#pragma unroll
  for (int nt = 0; nt < 8; ++nt) acc[nt] = vzero();

#pragma unroll
  for (int kt = 0; kt < K / 32; ++kt) {
    int k0 = kt * 32;
    v16h a = mk16(ld8(ap + k0 + akb), ld8(ap + k0 + akb + 16));
#pragma unroll
    for (int nt = 0; nt < 8; ++nt) {
      const _Float16* bp = Bt + (size_t)((ntOff + nt) * 16 + arow) * K + k0 + bkb;
      v16h b = mk16(ld8(bp), ld8(bp + 16));
      acc[nt] = __builtin_amdgcn_wmma_f32_16x16x32_f16(false, a, false, b,
                                                       (short)0, acc[nt], false, false);
    }
  }
  int cb = (lane < 16) ? 0 : 8;                 // C layout: vgpr r -> row cb+r, col arow
#pragma unroll
  for (int nt = 0; nt < 8; ++nt)
#pragma unroll
    for (int r = 0; r < 8; ++r)
      C[(size_t)(wave * 16 + cb + r) * ldC + (ntOff + nt) * 16 + arow] = acc[nt][r];
}

// ---------------- fused EdgeConv: m=[xi, xj-xi] -> relu(m@We1+be1)@We2+be2 -> max into dst
DEVINL h8 msg_chunk(const _Float16* pi, const _Float16* pj, int c) {
  if (c < 128) return ld8(pi + c);              // xi part
  return ld8(pj + c - 128) - ld8(pi + c - 128); // (xj - xi) part
}

__global__ void edgeconv_k(const _Float16* __restrict__ xh,
                           const int* __restrict__ src, const int* __restrict__ dst,
                           const _Float16* __restrict__ We1t, const float* __restrict__ be1,
                           const _Float16* __restrict__ We2t, const float* __restrict__ be2,
                           unsigned* __restrict__ aggE, int E) {
  __shared__ __align__(16) _Float16 lds[4][16 * 136];  // per-wave tile, padded pitch
  int wib  = threadIdx.x >> 5;
  int lane = threadIdx.x & 31;
  int tile = blockIdx.x * 4 + wib;
  int ebase = tile * 16;
  bool active = ebase < E;                      // wave-uniform
  int arow = lane & 15;
  int akb  = (lane < 16) ? 0 : 8;
  int bkb  = (lane < 16) ? 0 : 8;
  int cb   = (lane < 16) ? 0 : 8;

  v8f acc[8];
#pragma unroll
  for (int nt = 0; nt < 8; ++nt) acc[nt] = vzero();

  _Float16* L = lds[wib];

  if (active) {
    int e = ebase + arow; if (e >= E) e = E - 1;
    int s = src[e], d = dst[e];
    const _Float16* pi = xh + (size_t)d * 128;  // x_i = x[dst]
    const _Float16* pj = xh + (size_t)s * 128;  // x_j = x[src]
    __builtin_prefetch(pi, 0, 0);
    __builtin_prefetch(pj, 0, 0);
    // layer 1: [16 x 256] @ [256 x 128]
#pragma unroll
    for (int kt = 0; kt < 8; ++kt) {
      int k0 = kt * 32;
      v16h a = mk16(msg_chunk(pi, pj, k0 + akb), msg_chunk(pi, pj, k0 + akb + 16));
#pragma unroll
      for (int nt = 0; nt < 8; ++nt) {
        const _Float16* bp = We1t + (size_t)(nt * 16 + arow) * 256 + k0 + bkb;
        v16h b = mk16(ld8(bp), ld8(bp + 16));
        acc[nt] = __builtin_amdgcn_wmma_f32_16x16x32_f16(false, a, false, b,
                                                         (short)0, acc[nt], false, false);
      }
    }
    // bias + relu, stage through LDS to convert C-layout -> A-layout
#pragma unroll
    for (int nt = 0; nt < 8; ++nt) {
      float bias = be1[nt * 16 + arow];
#pragma unroll
      for (int r = 0; r < 8; ++r) {
        float v = acc[nt][r] + bias;
        L[(cb + r) * 136 + nt * 16 + arow] = (_Float16)fmaxf(v, 0.f);
      }
    }
  }
  __syncthreads();

#pragma unroll
  for (int nt = 0; nt < 8; ++nt) acc[nt] = vzero();

  if (active) {
    // layer 2: [16 x 128] @ [128 x 128]
    const _Float16* ap = L + arow * 136;
#pragma unroll
    for (int kt = 0; kt < 4; ++kt) {
      int k0 = kt * 32;
      v16h a = mk16(ld8(ap + k0 + akb), ld8(ap + k0 + akb + 16));
#pragma unroll
      for (int nt = 0; nt < 8; ++nt) {
        const _Float16* bp = We2t + (size_t)(nt * 16 + arow) * 128 + k0 + bkb;
        v16h b = mk16(ld8(bp), ld8(bp + 16));
        acc[nt] = __builtin_amdgcn_wmma_f32_16x16x32_f16(false, a, false, b,
                                                         (short)0, acc[nt], false, false);
      }
    }
    int drow[8]; bool dok[8];
#pragma unroll
    for (int r = 0; r < 8; ++r) {
      int ee = ebase + cb + r;
      dok[r]  = ee < E;
      drow[r] = dst[dok[r] ? ee : (E - 1)];
    }
#pragma unroll
    for (int nt = 0; nt < 8; ++nt) {
      float bias = be2[nt * 16 + arow];
#pragma unroll
      for (int r = 0; r < 8; ++r) {
        if (dok[r]) {
          float v = acc[nt][r] + bias;
          atomicMax(aggE + (size_t)drow[r] * 128 + nt * 16 + arow, encf(v));
        }
      }
    }
  }
}

// ---------------- small helper kernels -------------------------------------------------
__global__ void wcvt_k(const float* __restrict__ W, _Float16* __restrict__ Wt, int K, int Nn) {
  int i = blockIdx.x * blockDim.x + threadIdx.x;
  if (i >= K * Nn) return;
  int k = i / Nn, n = i % Nn;
  Wt[(size_t)n * K + k] = (_Float16)W[i];     // store transposed: [N, K]
}

__global__ void build_x0_k(const float* __restrict__ nodes, const int* __restrict__ cat,
                           const float* __restrict__ emb, _Float16* __restrict__ x0h) {
  int idx = blockIdx.x * blockDim.x + threadIdx.x;
  if (idx >= NN * 64) return;
  int i = idx >> 6, f = idx & 63;
  float v = (f < 32) ? nodes[(size_t)i * 32 + f] : emb[(size_t)cat[i] * 32 + (f - 32)];
  x0h[idx] = (_Float16)v;
}

__global__ void deg_init_k(float* deg) {
  int i = blockIdx.x * blockDim.x + threadIdx.x;
  if (i < NN) deg[i] = 1.0f;                  // self loop
}
__global__ void deg_edge_k(float* deg, const int* __restrict__ dst) {
  int e = blockIdx.x * blockDim.x + threadIdx.x;
  if (e < NE) atomicAdd(deg + dst[e], 1.0f);
}
__global__ void dinv_k(const float* __restrict__ deg, float* __restrict__ dinv) {
  int i = blockIdx.x * blockDim.x + threadIdx.x;
  if (i < NN) dinv[i] = rsqrtf(deg[i]);
}
__global__ void norm_k(const float* __restrict__ dinv, const int* __restrict__ src,
                       const int* __restrict__ dst, float* __restrict__ nrm) {
  int e = blockIdx.x * blockDim.x + threadIdx.x;
  if (e < NE) nrm[e] = dinv[src[e]] * dinv[dst[e]];
}

__global__ void agg_init_k(const float* __restrict__ h, const float* __restrict__ dinv,
                           float* __restrict__ agg) {
  int idx = blockIdx.x * blockDim.x + threadIdx.x;
  if (idx >= NN * 128) return;
  float di = dinv[idx >> 7];
  agg[idx] = di * di * h[idx];                // self-loop term, plain store
}
__global__ void agg_edge_k(const float* __restrict__ h, const int* __restrict__ src,
                           const int* __restrict__ dst, const float* __restrict__ nrm,
                           float* __restrict__ agg) {
  int idx = blockIdx.x * blockDim.x + threadIdx.x;
  if (idx >= NE * 32) return;
  int e = idx >> 5, q = (idx & 31) * 4;
  float w = nrm[e];
  float4 v = *(const float4*)(h + (size_t)src[e] * 128 + q);
  float* ap = agg + (size_t)dst[e] * 128 + q;
  atomicAdd(ap + 0, w * v.x); atomicAdd(ap + 1, w * v.y);
  atomicAdd(ap + 2, w * v.z); atomicAdd(ap + 3, w * v.w);
}

__global__ void bias_relu_h_k(const float* __restrict__ X, const float* __restrict__ b,
                              _Float16* __restrict__ Y, int n, int F) {
  int i = blockIdx.x * blockDim.x + threadIdx.x;
  if (i >= n) return;
  float v = X[i] + b[i % F];
  Y[i] = (_Float16)fmaxf(v, 0.f);
}
__global__ void bias_relu_f_k(const float* __restrict__ X, const float* __restrict__ b,
                              float* __restrict__ Y, int n, int F) {
  int i = blockIdx.x * blockDim.x + threadIdx.x;
  if (i >= n) return;
  float v = X[i] + b[i % F];
  Y[i] = fmaxf(v, 0.f);
}

__global__ void fill_u32_k(unsigned* p, unsigned v, int n) {
  int i = blockIdx.x * blockDim.x + threadIdx.x;
  if (i < n) p[i] = v;
}

__global__ void pool_accum_k(const unsigned* __restrict__ aggE, const int* __restrict__ batch,
                             unsigned* __restrict__ gmaxE, float* __restrict__ gsum,
                             float* __restrict__ cnt) {
  int idx = blockIdx.x * blockDim.x + threadIdx.x;
  if (idx >= NN * 32) return;
  int i = idx >> 5, q = (idx & 31) * 4;
  int g = batch[i];
#pragma unroll
  for (int t = 0; t < 4; ++t) {
    unsigned u = aggE[(size_t)i * 128 + q + t];   // enc(relu-max) directly
    atomicMax(gmaxE + (size_t)g * 128 + q + t, u);
    atomicAdd(gsum + (size_t)g * 128 + q + t, decf(u));
  }
  if (q == 0) atomicAdd(cnt + g, 1.0f);
}

__global__ void pool_fin_k(const unsigned* __restrict__ gmaxE, const float* __restrict__ gsum,
                           const float* __restrict__ cnt, _Float16* __restrict__ gh) {
  int idx = blockIdx.x * blockDim.x + threadIdx.x;
  if (idx >= NG * 256) return;
  int g = idx >> 8, f = idx & 255;
  float v = (f < 128) ? decf(gmaxE[(size_t)g * 128 + f])
                      : gsum[(size_t)g * 128 + (f - 128)] / fmaxf(cnt[g], 1.0f);
  gh[idx] = (_Float16)v;
}

// ---------------- host side -------------------------------------------------------------
extern "C" void kernel_launch(void* const* d_in, const int* in_sizes, int n_in,
                              void* d_out, int out_size, void* d_ws, size_t ws_size,
                              hipStream_t stream) {
  (void)in_sizes; (void)n_in; (void)out_size; (void)ws_size;
  const float* nodes = (const float*)d_in[0];
  const int*   cat   = (const int*)d_in[1];
  const int*   eidx  = (const int*)d_in[2];
  const int*   batch = (const int*)d_in[3];
  const float* emb   = (const float*)d_in[4];
  const float* W1 = (const float*)d_in[5];  const float* b1 = (const float*)d_in[6];
  const float* W2 = (const float*)d_in[7];  const float* b2 = (const float*)d_in[8];
  const float* W3 = (const float*)d_in[9];  const float* b3 = (const float*)d_in[10];
  const float* We1 = (const float*)d_in[11]; const float* be1 = (const float*)d_in[12];
  const float* We2 = (const float*)d_in[13]; const float* be2 = (const float*)d_in[14];
  const float* Wf1 = (const float*)d_in[15]; const float* bf1 = (const float*)d_in[16];
  const float* Wf2 = (const float*)d_in[17]; const float* bf2 = (const float*)d_in[18];
  float* out = (float*)d_out;
  const int* srcp = eidx;
  const int* dstp = eidx + NE;

  char* base = (char*)d_ws;
  size_t off = 0;
  auto take = [&](size_t bytes) -> char* {
    char* r = base + off;
    off = (off + bytes + 255) & ~(size_t)255;
    return r;
  };
  _Float16* x0h = (_Float16*)take((size_t)NN * 64 * 2);
  _Float16* xh  = (_Float16*)take((size_t)NN * 128 * 2);
  float*    h   = (float*)take((size_t)NN * 128 * 4);
  float*    agg = (float*)take((size_t)NN * 128 * 4);
  float*    deg = (float*)take((size_t)NN * 4);
  float*    dnv = (float*)take((size_t)NN * 4);
  float*    nrm = (float*)take((size_t)NE * 4);
  unsigned* gmx = (unsigned*)take((size_t)NG * 128 * 4);
  float*    gsm = (float*)take((size_t)NG * 128 * 4);
  float*    cnt = (float*)take((size_t)NG * 4);
  _Float16* gh  = (_Float16*)take((size_t)NG * 256 * 2);
  float*    hh  = (float*)take((size_t)NG * 256 * 4);
  _Float16* th  = (_Float16*)take((size_t)NG * 256 * 2);
  _Float16* w1t  = (_Float16*)take(64 * 128 * 2);
  _Float16* w2t  = (_Float16*)take(128 * 128 * 2);
  _Float16* w3t  = (_Float16*)take(128 * 128 * 2);
  _Float16* we1t = (_Float16*)take(256 * 128 * 2);
  _Float16* we2t = (_Float16*)take(128 * 128 * 2);
  _Float16* wf1t = (_Float16*)take(256 * 256 * 2);
  _Float16* wf2t = (_Float16*)take(256 * 256 * 2);

  auto cdiv = [](int a, int b) { return (a + b - 1) / b; };

  // weight transpose + f16 convert
  wcvt_k<<<cdiv(64 * 128, 256), 256, 0, stream>>>(W1, w1t, 64, 128);
  wcvt_k<<<cdiv(128 * 128, 256), 256, 0, stream>>>(W2, w2t, 128, 128);
  wcvt_k<<<cdiv(128 * 128, 256), 256, 0, stream>>>(W3, w3t, 128, 128);
  wcvt_k<<<cdiv(256 * 128, 256), 256, 0, stream>>>(We1, we1t, 256, 128);
  wcvt_k<<<cdiv(128 * 128, 256), 256, 0, stream>>>(We2, we2t, 128, 128);
  wcvt_k<<<cdiv(256 * 256, 256), 256, 0, stream>>>(Wf1, wf1t, 256, 256);
  wcvt_k<<<cdiv(256 * 256, 256), 256, 0, stream>>>(Wf2, wf2t, 256, 256);

  // features + GCN normalization
  build_x0_k<<<cdiv(NN * 64, 256), 256, 0, stream>>>(nodes, cat, emb, x0h);
  deg_init_k<<<cdiv(NN, 256), 256, 0, stream>>>(deg);
  deg_edge_k<<<cdiv(NE, 256), 256, 0, stream>>>(deg, dstp);
  dinv_k<<<cdiv(NN, 256), 256, 0, stream>>>(deg, dnv);
  norm_k<<<cdiv(NE, 256), 256, 0, stream>>>(dnv, srcp, dstp, nrm);

  const int rowTiles = NN / 16;  // 3125
  dim3 gemmGrid(cdiv(rowTiles, 4), 1);

  // GCN layer 1
  gemm_f16<64><<<gemmGrid, 128, 0, stream>>>(x0h, w1t, h, rowTiles, 128);
  agg_init_k<<<cdiv(NN * 128, 256), 256, 0, stream>>>(h, dnv, agg);
  agg_edge_k<<<cdiv(NE * 32, 256), 256, 0, stream>>>(h, srcp, dstp, nrm, agg);
  bias_relu_h_k<<<cdiv(NN * 128, 256), 256, 0, stream>>>(agg, b1, xh, NN * 128, 128);
  // GCN layer 2
  gemm_f16<128><<<gemmGrid, 128, 0, stream>>>(xh, w2t, h, rowTiles, 128);
  agg_init_k<<<cdiv(NN * 128, 256), 256, 0, stream>>>(h, dnv, agg);
  agg_edge_k<<<cdiv(NE * 32, 256), 256, 0, stream>>>(h, srcp, dstp, nrm, agg);
  bias_relu_h_k<<<cdiv(NN * 128, 256), 256, 0, stream>>>(agg, b2, xh, NN * 128, 128);
  // GCN layer 3
  gemm_f16<128><<<gemmGrid, 128, 0, stream>>>(xh, w3t, h, rowTiles, 128);
  agg_init_k<<<cdiv(NN * 128, 256), 256, 0, stream>>>(h, dnv, agg);
  agg_edge_k<<<cdiv(NE * 32, 256), 256, 0, stream>>>(h, srcp, dstp, nrm, agg);
  bias_relu_h_k<<<cdiv(NN * 128, 256), 256, 0, stream>>>(agg, b3, xh, NN * 128, 128);

  // EdgeConv (fused, WMMA) -> encoded max-aggregation; init enc(0) folds relu + isolated-node rule
  fill_u32_k<<<cdiv(NN * 128, 256), 256, 0, stream>>>((unsigned*)agg, 0x80000000u, NN * 128);
  edgeconv_k<<<cdiv(cdiv(NE, 16), 4), 128, 0, stream>>>(xh, srcp, dstp, we1t, be1,
                                                        we2t, be2, (unsigned*)agg, NE);

  // global max || mean pool
  fill_u32_k<<<cdiv(NG * 128, 256), 256, 0, stream>>>(gmx, 0x80000000u, NG * 128);
  fill_u32_k<<<cdiv(NG * 128, 256), 256, 0, stream>>>((unsigned*)gsm, 0u, NG * 128);
  fill_u32_k<<<1, 64, 0, stream>>>((unsigned*)cnt, 0u, NG);
  pool_accum_k<<<cdiv(NN * 32, 256), 256, 0, stream>>>((unsigned*)agg, batch, gmx, gsm, cnt);
  pool_fin_k<<<cdiv(NG * 256, 256), 256, 0, stream>>>(gmx, gsm, cnt, gh);

  // head MLP (two WMMA GEMMs, M=64)
  dim3 headGrid(1, 2);
  gemm_f16<256><<<headGrid, 128, 0, stream>>>(gh, wf1t, hh, 4, 256);
  bias_relu_h_k<<<cdiv(NG * 256, 256), 256, 0, stream>>>(hh, bf1, th, NG * 256, 256);
  gemm_f16<256><<<headGrid, 128, 0, stream>>>(th, wf2t, hh, 4, 256);
  bias_relu_f_k<<<cdiv(NG * 256, 256), 256, 0, stream>>>(hh, bf2, out, NG * 256, 256);
}
